// Attention_26371099197664
// MI455X (gfx1250) — matrix operand order repaired and checked by reading, not compiled, not run on hardware
//
#include <hip/hip_runtime.h>
#include <math.h>

// ---------------- types for CDNA5 WMMA ----------------
typedef __attribute__((ext_vector_type(16))) __bf16 v16bf;
typedef __attribute__((ext_vector_type(8)))  __bf16 v8bf;
typedef __attribute__((ext_vector_type(8)))  float  v8f;

// gcc-style int4 to match the async-to-LDS builtin's parameter type
typedef int v4i_g __attribute__((vector_size(16)));
typedef __attribute__((address_space(1))) v4i_g as1_v4i;
typedef __attribute__((address_space(3))) v4i_g as3_v4i;

// ---------------- problem constants ----------------
#define BATCH   4
#define NCTX    2048
#define TOK     (BATCH * NCTX)     // 8192
#define DIM     512
#define HEADS   8
#define DH      64
#define NQKV    (3 * DIM)          // 1536 (Q | K | V columns)
#define ATTN_SCALE 0.125f          // 64^-0.5

// ---------------- CDNA5 async global->LDS (ASYNCcnt path) ----------------
#if defined(__HIP_DEVICE_COMPILE__) && __has_builtin(__builtin_amdgcn_global_load_async_to_lds_b128)
#define HAVE_ASYNC_LDS 1
#else
#define HAVE_ASYNC_LDS 0
#endif

__device__ __forceinline__ void async_copy16(unsigned short* lds, const unsigned short* g) {
#if HAVE_ASYNC_LDS
    __builtin_amdgcn_global_load_async_to_lds_b128(
        (as1_v4i*)g, (as3_v4i*)lds, 0, 0);
#else
    *(v8bf*)lds = *(const v8bf*)g;          // sync fallback
#endif
}

template<int N> __device__ __forceinline__ void wait_async() {
#if HAVE_ASYNC_LDS
#if __has_builtin(__builtin_amdgcn_s_wait_asynccnt)
    __builtin_amdgcn_s_wait_asynccnt((unsigned short)N);
#else
    asm volatile("s_wait_asynccnt %0" :: "i"(N) : "memory");
#endif
#endif
}

__device__ __forceinline__ unsigned short f2bf_rn(float f) {
    union { float f; unsigned u; } v; v.f = f;
    unsigned r = v.u + 0x7FFFu + ((v.u >> 16) & 1u);   // round-to-nearest-even
    return (unsigned short)(r >> 16);
}

// ---------------- conversion kernels ----------------
__global__ void cvt_x_kernel(const float* __restrict__ x, unsigned short* __restrict__ xb, int n) {
    int i = blockIdx.x * 256 + threadIdx.x;
    if (i < n) xb[i] = f2bf_rn(x[i]);
}

// Builds WqkvT [1536,512] (row n = output col, contiguous K) and WoT [512,512].
__global__ void cvt_w_kernel(const float* __restrict__ Wq, const float* __restrict__ Wkv,
                             const float* __restrict__ Wo,
                             unsigned short* __restrict__ WqkvT, unsigned short* __restrict__ WoT) {
    int i = blockIdx.x * 256 + threadIdx.x;
    if (i < NQKV * DIM) {
        int nn = i >> 9, k = i & 511;
        float v = (nn < DIM) ? Wq[(size_t)k * DIM + nn]
                             : Wkv[(size_t)k * (2 * DIM) + (nn - DIM)];
        WqkvT[i] = f2bf_rn(v);
    } else {
        int j = i - NQKV * DIM;                 // < 512*512
        int nn = j >> 9, k = j & 511;
        WoT[j] = f2bf_rn(Wo[(size_t)k * DIM + nn]);
    }
}

// ---------------- pipelined 128x128 workgroup GEMM ----------------
// 256 threads = 8 waves in a 4(M) x 2(N) grid; each wave owns a 32x64 tile.
// K=512 consumed in 16 fully-unrolled steps of 32, double-buffered through LDS
// via global_load_async_to_lds_b128 (ASYNCcnt) so WMMAs overlap the next stage.
// MODE 0: C = xb @ WqkvT^T, scatter epilogue into Qb / Kb / Vt (bf16)
// MODE 1: C = Ob @ WoT^T + bias, f32 epilogue into out
template<int MODE>
__global__ __launch_bounds__(256) void gemm128_kernel(
        const unsigned short* __restrict__ A,   // [M, 512] bf16
        const unsigned short* __restrict__ BT,  // [N, 512] bf16 (pre-transposed weights)
        const float* __restrict__ bias,
        unsigned short* __restrict__ Qb, unsigned short* __restrict__ Kb,
        unsigned short* __restrict__ Vt,
        float* __restrict__ out,
        int ntiles_n) {
    __shared__ unsigned short lA[2][128 * 32];   // 8 KB per buffer
    __shared__ unsigned short lB[2][128 * 32];

    const int tid  = threadIdx.x;
    const int lane = tid & 31, wave = tid >> 5;
    const int hh   = lane >> 4, mcol = lane & 15;
    const int wm   = wave >> 1, wn = wave & 1;
    const int tileM = blockIdx.x / ntiles_n, tileN = blockIdx.x % ntiles_n;

    const unsigned short* Ablk = A  + (size_t)tileM * 128 * DIM;
    const unsigned short* Bblk = BT + (size_t)tileN * 128 * DIM;

    v8f acc[2][4] = {};

    // stage kt: copy A/B 128x32 k-slices into LDS buffer `buf`.
    // Each tile = 512 chunks of 16B; 256 threads x 2 chunks; 4 async ops/wave/stage.
#define STAGE(KT, BUF)                                                              \
    {                                                                               \
        int k0_ = (KT) * 32;                                                        \
        _Pragma("unroll")                                                           \
        for (int i_ = 0; i_ < 2; ++i_) {                                            \
            int ch_ = tid + i_ * 256;                                               \
            int row_ = ch_ >> 2, c8_ = (ch_ & 3) * 8;                               \
            async_copy16(&lA[BUF][row_ * 32 + c8_],                                 \
                         Ablk + (size_t)row_ * DIM + k0_ + c8_);                    \
            async_copy16(&lB[BUF][row_ * 32 + c8_],                                 \
                         Bblk + (size_t)row_ * DIM + k0_ + c8_);                    \
        }                                                                           \
    }

    STAGE(0, 0);
    const int NK = DIM / 32;                     // 16
#pragma unroll
    for (int kt = 0; kt < NK; ++kt) {
        const int buf = kt & 1;
        if (kt + 1 < NK) { STAGE(kt + 1, buf ^ 1); wait_async<4>(); }
        else             { wait_async<0>(); }
        __syncthreads();

        const unsigned short* la = lA[buf];
        const unsigned short* lb = lB[buf];
        v16bf a[2];
#pragma unroll
        for (int s = 0; s < 2; ++s) {
            int row = wm * 32 + s * 16 + mcol;
            v8bf lo = *(const v8bf*)(la + row * 32 + hh * 16);
            v8bf hi = *(const v8bf*)(la + row * 32 + hh * 16 + 8);
            a[s] = __builtin_shufflevector(lo, hi, 0,1,2,3,4,5,6,7,8,9,10,11,12,13,14,15);
        }
#pragma unroll
        for (int nt = 0; nt < 4; ++nt) {
            int brow = wn * 64 + nt * 16 + mcol;
            v16bf b = *(const v16bf*)(lb + brow * 32 + hh * 16);
            acc[0][nt] = __builtin_amdgcn_wmma_f32_16x16x32_bf16(
                false, a[0], false, b, (short)0, acc[0][nt], false, false);
            acc[1][nt] = __builtin_amdgcn_wmma_f32_16x16x32_bf16(
                false, a[1], false, b, (short)0, acc[1][nt], false, false);
        }
        __syncthreads();
    }
#undef STAGE

#pragma unroll
    for (int s = 0; s < 2; ++s)
#pragma unroll
        for (int nt = 0; nt < 4; ++nt) {
            int col = tileN * 128 + wn * 64 + nt * 16 + mcol;
#pragma unroll
            for (int r = 0; r < 8; ++r) {
                int tok = tileM * 128 + wm * 32 + s * 16 + r + hh * 8;
                if (MODE == 0) {
                    unsigned short val = f2bf_rn(acc[s][nt][r]);
                    int seg = col >> 9, h = (col >> 6) & 7, d = col & 63;
                    if (seg == 0)      Qb[(size_t)tok * DIM + h * DH + d] = val;
                    else if (seg == 1) Kb[(size_t)tok * DIM + h * DH + d] = val;
                    else {
                        int bb = tok >> 11, nr = tok & (NCTX - 1);
                        Vt[(((size_t)(bb * HEADS + h)) * DH + d) * NCTX + nr] = val;
                    }
                } else {
                    out[(size_t)tok * DIM + col] = acc[s][nt][r] + bias[col];
                }
            }
        }
}

// ---------------- flash attention ----------------
// 4 waves per block share one (batch, head): 64 query rows; K/V tiles for 32
// keys are async-staged into double-buffered LDS once per block per step.
__global__ __launch_bounds__(128) void flash_attn_kernel(
        const unsigned short* __restrict__ Qb, const unsigned short* __restrict__ Kb,
        const unsigned short* __restrict__ Vt, unsigned short* __restrict__ Ob) {
    __shared__ unsigned short lK[2][32 * 64];     // 4 KB each: [key(32)][d(64)]
    __shared__ unsigned short lV[2][64 * 32];     // 4 KB each: [d(64)][key(32)]
    __shared__ unsigned short plds[4][16 * 32];   // per-wave P C->A staging

    const int tid  = threadIdx.x;
    const int lane = tid & 31, wave = tid >> 5;
    const int hh   = lane >> 4, mcol = lane & 15;

    int bh = blockIdx.x >> 5;                     // 32 q-chunks of 64 rows per bh
    int q0 = (blockIdx.x & 31) * 64 + wave * 16;
    int bb = bh >> 3, h = bh & 7;

    // Q tile in A layout (contraction over d, two 32-wide chunks)
    const unsigned short* qrow = Qb + (size_t)(bb * NCTX + q0 + mcol) * DIM + h * DH;
    v16bf qa[2];
#pragma unroll
    for (int dd = 0; dd < 2; ++dd) {
        v8bf lo = *(const v8bf*)(qrow + dd * 32 + hh * 16);
        v8bf hi = *(const v8bf*)(qrow + dd * 32 + hh * 16 + 8);
        qa[dd] = __builtin_shufflevector(lo, hi, 0,1,2,3,4,5,6,7,8,9,10,11,12,13,14,15);
    }

    v8f acc[4] = {};
    float mrun[8], lrun[8];
#pragma unroll
    for (int r = 0; r < 8; ++r) { mrun[r] = -1e30f; lrun[r] = 0.f; }

    const unsigned short* kbase = Kb + (size_t)(bb * NCTX) * DIM + h * DH;   // row j stride 512
    const unsigned short* vbase = Vt + (size_t)bh * DH * NCTX;               // row d stride 2048
    unsigned short* pl = plds[wave];

    // K tile: 32 rows x 128B = 256 chunks; V tile: 64 rows x 64B = 256 chunks.
    // 128 threads x 2 chunks each; 4 async ops/wave/stage.
#define STAGE_KV(J0, BUF)                                                           \
    {                                                                               \
        _Pragma("unroll")                                                           \
        for (int i_ = 0; i_ < 2; ++i_) {                                            \
            int ch_ = tid + i_ * 128;                                               \
            async_copy16(&lK[BUF][(ch_ >> 3) * 64 + (ch_ & 7) * 8],                 \
                         kbase + (size_t)((J0) + (ch_ >> 3)) * DIM + (ch_ & 7) * 8);\
            async_copy16(&lV[BUF][(ch_ >> 2) * 32 + (ch_ & 3) * 8],                 \
                         vbase + (size_t)(ch_ >> 2) * NCTX + (J0) + (ch_ & 3) * 8); \
        }                                                                           \
    }

    STAGE_KV(0, 0);
#pragma unroll 2
    for (int j0 = 0; j0 < NCTX; j0 += 32) {
        const int buf = (j0 >> 5) & 1;
        if (j0 + 32 < NCTX) { STAGE_KV(j0 + 32, buf ^ 1); wait_async<4>(); }
        else                { wait_async<0>(); }
        __syncthreads();

        // S = Q K^T  (16 x 32 tile as two 16x16 accumulators), K frags from LDS
        v8f s0 = {}, s1 = {};
#pragma unroll
        for (int dd = 0; dd < 2; ++dd) {
            v16bf kb0 = *(const v16bf*)(&lK[buf][(mcol)      * 64 + dd * 32 + hh * 16]);
            v16bf kb1 = *(const v16bf*)(&lK[buf][(16 + mcol) * 64 + dd * 32 + hh * 16]);
            s0 = __builtin_amdgcn_wmma_f32_16x16x32_bf16(false, qa[dd], false, kb0, (short)0, s0, false, false);
            s1 = __builtin_amdgcn_wmma_f32_16x16x32_bf16(false, qa[dd], false, kb1, (short)0, s1, false, false);
        }

        // online softmax (row r+8*hh lives in slot r; N spans the 16-lane half)
        float rmax[8], alpha[8], psum[8];
#pragma unroll
        for (int r = 0; r < 8; ++r) {
            float a0 = s0[r] * ATTN_SCALE, a1 = s1[r] * ATTN_SCALE;
            s0[r] = a0; s1[r] = a1;
            rmax[r] = fmaxf(a0, a1);
        }
#pragma unroll
        for (int off = 1; off < 16; off <<= 1) {
#pragma unroll
            for (int r = 0; r < 8; ++r)
                rmax[r] = fmaxf(rmax[r], __shfl_xor(rmax[r], off, 32));
        }
#pragma unroll
        for (int r = 0; r < 8; ++r) {
            float mn = fmaxf(mrun[r], rmax[r]);
            alpha[r] = __expf(mrun[r] - mn);
            mrun[r] = mn;
            float p0 = __expf(s0[r] - mn);
            float p1 = __expf(s1[r] - mn);
            psum[r] = p0 + p1;
            pl[(r + hh * 8) * 32 + mcol]      = f2bf_rn(p0);
            pl[(r + hh * 8) * 32 + 16 + mcol] = f2bf_rn(p1);
        }
#pragma unroll
        for (int off = 1; off < 16; off <<= 1) {
#pragma unroll
            for (int r = 0; r < 8; ++r)
                psum[r] += __shfl_xor(psum[r], off, 32);
        }
#pragma unroll
        for (int r = 0; r < 8; ++r) lrun[r] = lrun[r] * alpha[r] + psum[r];
#pragma unroll
        for (int nt = 0; nt < 4; ++nt)
#pragma unroll
            for (int r = 0; r < 8; ++r) acc[nt][r] *= alpha[r];

        // C-layout -> A-layout through per-wave LDS (CDNA5 split DS counter)
        asm volatile("s_wait_dscnt 0" ::: "memory");
        v8bf plo = *(const v8bf*)(pl + mcol * 32 + hh * 16);
        v8bf phi = *(const v8bf*)(pl + mcol * 32 + hh * 16 + 8);
        v16bf pa = __builtin_shufflevector(plo, phi, 0,1,2,3,4,5,6,7,8,9,10,11,12,13,14,15);

        // O += P V  (contraction over 32 keys), V frags from LDS
#pragma unroll
        for (int nt = 0; nt < 4; ++nt) {
            v16bf vb = *(const v16bf*)(&lV[buf][(nt * 16 + mcol) * 32 + hh * 16]);
            acc[nt] = __builtin_amdgcn_wmma_f32_16x16x32_bf16(
                false, pa, false, vb, (short)0, acc[nt], false, false);
        }
        __syncthreads();
    }
#undef STAGE_KV

    // normalize + store O as [tok, h*64+d] bf16
    unsigned short* orow = Ob + (size_t)(bb * NCTX + q0) * DIM + h * DH;
    float inv[8];
#pragma unroll
    for (int r = 0; r < 8; ++r) inv[r] = 1.f / lrun[r];
#pragma unroll
    for (int nt = 0; nt < 4; ++nt)
#pragma unroll
        for (int r = 0; r < 8; ++r)
            orow[(size_t)(r + hh * 8) * DIM + nt * 16 + mcol] = f2bf_rn(acc[nt][r] * inv[r]);
}

// ---------------- host launch ----------------
extern "C" void kernel_launch(void* const* d_in, const int* in_sizes, int n_in,
                              void* d_out, int out_size, void* d_ws, size_t ws_size,
                              hipStream_t stream) {
    const float* x   = (const float*)d_in[0];
    const float* Wq  = (const float*)d_in[1];
    const float* Wkv = (const float*)d_in[2];
    const float* Wo  = (const float*)d_in[3];
    const float* bo  = (const float*)d_in[4];
    float* out = (float*)d_out;

    char* ws = (char*)d_ws;
    size_t off = 0;
    unsigned short* xb    = (unsigned short*)(ws + off); off += (size_t)TOK * DIM * 2;
    unsigned short* WqkvT = (unsigned short*)(ws + off); off += (size_t)NQKV * DIM * 2;
    unsigned short* WoT   = (unsigned short*)(ws + off); off += (size_t)DIM * DIM * 2;
    unsigned short* Qb    = (unsigned short*)(ws + off); off += (size_t)TOK * DIM * 2;
    unsigned short* Kb    = (unsigned short*)(ws + off); off += (size_t)TOK * DIM * 2;
    unsigned short* Vt    = (unsigned short*)(ws + off); off += (size_t)TOK * DIM * 2;
    unsigned short* Ob    = (unsigned short*)(ws + off); off += (size_t)TOK * DIM * 2;

    cvt_x_kernel<<<(TOK * DIM + 255) / 256, 256, 0, stream>>>(x, xb, TOK * DIM);
    cvt_w_kernel<<<(NQKV * DIM + DIM * DIM + 255) / 256, 256, 0, stream>>>(Wq, Wkv, Wo, WqkvT, WoT);
    gemm128_kernel<0><<<(TOK / 128) * (NQKV / 128), 256, 0, stream>>>(
        xb, WqkvT, nullptr, Qb, Kb, Vt, nullptr, NQKV / 128);
    flash_attn_kernel<<<(BATCH * HEADS) * (NCTX / 64), 128, 0, stream>>>(Qb, Kb, Vt, Ob);
    gemm128_kernel<1><<<(TOK / 128) * (DIM / 128), 256, 0, stream>>>(
        Ob, WoT, bo, nullptr, nullptr, nullptr, out, DIM / 128);
}
